// GATModelVAE_71725953843275
// MI455X (gfx1250) — matrix operand order repaired and checked by reading, not compiled, not run on hardware
//
#include <hip/hip_runtime.h>
#include <hip/hip_bf16.h>

// ---------------------------------------------------------------------------
// GAT-VAE forward for MI455X (gfx1250, wave32, WMMA).
//   N=50000 nodes, E=1600000 edges, F_IN=512, H1=128, H2=64
//   out = [z (=mu), mu, logvar] each [N, 64] fp32, concatenated flat.
// Dense GEMMs: v_wmma_f32_16x16x32_bf16 (bf16 in, f32 accum), B pre-swizzled
// in LDS to fragment order (wide ds_load_b128 reads), A via global b128 loads
// with in-register f32->bf16 conversion. Edge softmax/aggregation: native f32
// global atomics (bandwidth/L2 bound).
// ---------------------------------------------------------------------------

#define NN      50000
#define NE      1600000
#define F_IN    512
#define H1D     128
#define H2D     64
#define SLOPE   0.01f
#define INV_T   2.0f              // 1/TEMP, TEMP=0.5
#define NEG_INF_ENC 0x007FFFFFu   // ordered-uint encoding of -inf

typedef __attribute__((ext_vector_type(16))) __bf16 v16bf;
typedef __attribute__((ext_vector_type(8)))  float  v8f;

// ------------------------------- utilities ---------------------------------

__global__ void fill_f32(float* __restrict__ p, float v, int n) {
    int i = blockIdx.x * blockDim.x + threadIdx.x;
    if (i < n) p[i] = v;
}
__global__ void fill_u32(unsigned* __restrict__ p, unsigned v, int n) {
    int i = blockIdx.x * blockDim.x + threadIdx.x;
    if (i < n) p[i] = v;
}
__global__ void copy_f32(const float* __restrict__ s, float* __restrict__ d, int n) {
    int i = blockIdx.x * blockDim.x + threadIdx.x;
    if (i < n) d[i] = s[i];
}
__global__ void cvt_f32_bf16(const float* __restrict__ s, __bf16* __restrict__ d, int n) {
    int i = blockIdx.x * blockDim.x + threadIdx.x;
    if (i < n) d[i] = (__bf16)s[i];
}

// Order-preserving float -> uint encoding (atomic max over signed floats)
__device__ __forceinline__ unsigned ordEnc(float f) {
    unsigned u = __float_as_uint(f);
    return (u & 0x80000000u) ? ~u : (u | 0x80000000u);
}
__device__ __forceinline__ float ordDec(unsigned d) {
    unsigned u = (d & 0x80000000u) ? (d & 0x7FFFFFFFu) : ~d;
    return __uint_as_float(u);
}

// ------------------------------ WMMA GEMM ----------------------------------
// C[M,NC] (f32) = A[M,K] (f32, optional ReLU) x B[K,NC] (bf16)
// block = 256 threads = 8 waves; each wave -> one 16x16 output tile;
// block covers 128 rows x 16 cols.
//
// LDS holds B pre-swizzled into WMMA fragment order:
//   Bsw[kblk][lane][16] bf16, where fragment element j of lane L at k-block
//   kblk is B[kblk*32 + (L>>4)*16 + j, colBase + (L&15)].
// Each lane then reads its whole 32-byte fragment with aligned wide loads.
template <bool RELU_A, int NC>
__global__ void wmma_gemm_bf16(const float*  __restrict__ A,
                               const __bf16* __restrict__ B,
                               float*        __restrict__ C,
                               int M, int K)
{
    extern __shared__ __bf16 Bsw[];           // [K/32][32][16]
    const int colBase = blockIdx.y * 16;

    // Cooperative, globally-coalesced load of the 16-column B strip,
    // scattered into fragment order in LDS.
    for (int idx = threadIdx.x; idx < K * 16; idx += blockDim.x) {
        const int k   = idx >> 4;             // 0..K-1
        const int col = idx & 15;
        const int kblk   = k >> 5;
        const int within = k & 31;
        const int hi = within >> 4;           // which half-wave owns this k
        const int j  = within & 15;           // fragment element index
        Bsw[(kblk << 9) + ((hi << 4) + col) * 16 + j] = B[k * NC + colBase + col];
    }
    __syncthreads();

    const int wave = threadIdx.x >> 5;
    const int lane = threadIdx.x & 31;
    const int hi   = lane >> 4;               // 0 / 1 half-wave
    const int l16  = lane & 15;
    const int rowBase = blockIdx.x * 128 + wave * 16;
    const int arow    = rowBase + l16;
    // Clamp address only; garbage rows feed only C rows that are never stored.
    const float* aRow = A + (size_t)(arow < M ? arow : (M - 1)) * K;
    const __bf16* bFragBase = Bsw + lane * 16;

    v8f acc = {};
    for (int kb = 0; kb < K; kb += 32) {
        // A fragment (contiguous, matching B swizzle): elements 0..15 = K [kb+hi*16 .. +15]
        //   -> four b128 loads.
        const float4 q0 = *(const float4*)(aRow + kb + hi * 16);
        const float4 q1 = *(const float4*)(aRow + kb + hi * 16 + 4);
        const float4 q2 = *(const float4*)(aRow + kb + hi * 16 + 8);
        const float4 q3 = *(const float4*)(aRow + kb + hi * 16 + 12);
        float af[16] = { q0.x, q0.y, q0.z, q0.w, q1.x, q1.y, q1.z, q1.w,
                         q2.x, q2.y, q2.z, q2.w, q3.x, q3.y, q3.z, q3.w };
        v16bf afrag;
        #pragma unroll
        for (int i = 0; i < 16; ++i) {
            float v = RELU_A ? fmaxf(af[i], 0.f) : af[i];
            afrag[i] = (__bf16)v;
        }
        // B fragment: single aligned 32-byte LDS read in fragment order.
        const v16bf bfrag = *(const v16bf*)(bFragBase + (kb >> 5) * 512);

        acc = __builtin_amdgcn_wmma_f32_16x16x32_bf16(
                  false, afrag, false, bfrag, (short)0, acc, false, false);
    }

    // C layout: VGPR r, lanes 0-15 -> M = r, lanes 16-31 -> M = 8 + r; N = l16.
    float* cBase = C + (size_t)(rowBase + hi * 8) * NC + colBase + l16;
    if (rowBase + 16 <= M) {                   // full tile fast path
        #pragma unroll
        for (int r = 0; r < 8; ++r) cBase[r * NC] = acc[r];
    } else {
        #pragma unroll
        for (int r = 0; r < 8; ++r)
            if (rowBase + hi * 8 + r < M) cBase[r * NC] = acc[r];
    }
}

// ----------------------- per-node attention projections --------------------
// s[i] = Wh[i,:] . a[0:D] ; t[i] = Wh[i,:] . a[D:2D]   (wave per node)
template <int D>
__global__ void node_st(const float* __restrict__ Wh, const float* __restrict__ a,
                        float* __restrict__ s, float* __restrict__ t, int n)
{
    constexpr int VEC = D / 32;               // 4 (D=128) or 2 (D=64)
    int w    = (blockIdx.x * blockDim.x + threadIdx.x) >> 5;
    int lane = threadIdx.x & 31;
    if (w >= n) return;
    const float* row = Wh + (size_t)w * D + lane * VEC;
    const float* as  = a + lane * VEC;
    const float* at  = a + D + lane * VEC;
    float sv = 0.f, tv = 0.f;
    if constexpr (VEC == 4) {
        float4 x  = *(const float4*)row;
        float4 c0 = *(const float4*)as;
        float4 c1 = *(const float4*)at;
        sv = x.x * c0.x + x.y * c0.y + x.z * c0.z + x.w * c0.w;
        tv = x.x * c1.x + x.y * c1.y + x.z * c1.z + x.w * c1.w;
    } else {
        float2 x  = *(const float2*)row;
        float2 c0 = *(const float2*)as;
        float2 c1 = *(const float2*)at;
        sv = x.x * c0.x + x.y * c0.y;
        tv = x.x * c1.x + x.y * c1.y;
    }
    #pragma unroll
    for (int off = 16; off > 0; off >>= 1) {
        sv += __shfl_xor(sv, off, 32);
        tv += __shfl_xor(tv, off, 32);
    }
    if (lane == 0) { s[w] = sv; t[w] = tv; }
}

// ------------------------------- edge passes -------------------------------

__global__ void edge_logits_max(const int* __restrict__ src, const int* __restrict__ dst,
                                const float* __restrict__ s, const float* __restrict__ t,
                                float* __restrict__ ew, unsigned* __restrict__ menc, int ne)
{
    int e = blockIdx.x * blockDim.x + threadIdx.x;
    if (e >= ne) return;
    int sr = src[e];
    float v = s[sr] + t[dst[e]];
    v = (v > 0.f ? v : SLOPE * v) * INV_T;
    ew[e] = v;
    atomicMax(&menc[sr], ordEnc(v));
}

__global__ void node_max_fix(const unsigned* __restrict__ menc, float* __restrict__ m, int n)
{
    int i = blockIdx.x * blockDim.x + threadIdx.x;
    if (i >= n) return;
    float v = ordDec(menc[i]);
    if (v == -__builtin_inff()) v = 0.f;      // guard empty rows (matches reference)
    m[i] = v;
}

__global__ void edge_exp_sum(const int* __restrict__ src, const float* __restrict__ m,
                             float* __restrict__ ew, float* __restrict__ denom, int ne)
{
    int e = blockIdx.x * blockDim.x + threadIdx.x;
    if (e >= ne) return;
    int sr = src[e];
    float ex = expf(ew[e] - m[sr]);
    ew[e] = ex;
    atomicAdd(&denom[sr], ex);
}

// hp[src,:] += (ew/denom[src]) * Wh[dst,:]   — one wave per edge.
template <int D>
__global__ void edge_scatter(const int* __restrict__ src, const int* __restrict__ dst,
                             const float* __restrict__ ew, const float* __restrict__ denom,
                             const float* __restrict__ Wh, float* __restrict__ hp, int ne)
{
    constexpr int VEC = D / 32;               // 4 (D=128) or 2 (D=64)
    int e    = (blockIdx.x * blockDim.x + threadIdx.x) >> 5;
    int lane = threadIdx.x & 31;
    if (e >= ne) return;
    int sr = src[e], dr = dst[e];
    float att = ew[e] / denom[sr];
    const float* w = Wh + (size_t)dr * D + lane * VEC;
    float*       o = hp + (size_t)sr * D + lane * VEC;
    if constexpr (VEC == 4) {
        float4 v = *(const float4*)w;
        atomicAdd(o + 0, att * v.x);
        atomicAdd(o + 1, att * v.y);
        atomicAdd(o + 2, att * v.z);
        atomicAdd(o + 3, att * v.w);
    } else {
        float2 v = *(const float2*)w;
        atomicAdd(o + 0, att * v.x);
        atomicAdd(o + 1, att * v.y);
    }
}

// ------------------------------- host driver -------------------------------

static inline size_t alignUp(size_t v) { return (v + 255) & ~(size_t)255; }

extern "C" void kernel_launch(void* const* d_in, const int* in_sizes, int n_in,
                              void* d_out, int out_size, void* d_ws, size_t ws_size,
                              hipStream_t stream)
{
    const float* x  = (const float*)d_in[0];
    const float* W1 = (const float*)d_in[1];
    const float* a1 = (const float*)d_in[2];
    const float* W2 = (const float*)d_in[3];
    const float* a2 = (const float*)d_in[4];
    const float* W3 = (const float*)d_in[5];
    const float* a3 = (const float*)d_in[6];
    const int*   ei = (const int*)d_in[7];
    const int* src = ei;          // edge_index[0]
    const int* dst = ei + NE;     // edge_index[1]
    float* out = (float*)d_out;

    // ---- workspace layout (bytes) ----
    char* ws = (char*)d_ws;
    size_t o = 0;
    auto take = [&](size_t bytes) { size_t r = o; o += alignUp(bytes); return r; };
    __bf16* W1b   = (__bf16*)(ws + take((size_t)F_IN * H1D * sizeof(__bf16)));
    __bf16* W2b   = (__bf16*)(ws + take((size_t)H1D * H2D * sizeof(__bf16)));
    __bf16* W3b   = (__bf16*)(ws + take((size_t)H1D * H2D * sizeof(__bf16)));
    float*  Wh1   = (float*) (ws + take((size_t)NN * H1D * sizeof(float)));
    float*  hp1   = (float*) (ws + take((size_t)NN * H1D * sizeof(float)));
    float*  Wh2   = (float*) (ws + take((size_t)NN * H2D * sizeof(float)));
    float*  Wh3   = (float*) (ws + take((size_t)NN * H2D * sizeof(float)));
    float*  sBuf  = (float*) (ws + take((size_t)NN * sizeof(float)));
    float*  tBuf  = (float*) (ws + take((size_t)NN * sizeof(float)));
    unsigned* menc= (unsigned*)(ws + take((size_t)NN * sizeof(unsigned)));
    float*  mBuf  = (float*) (ws + take((size_t)NN * sizeof(float)));
    float*  denom = (float*) (ws + take((size_t)NN * sizeof(float)));
    float*  ework = (float*) (ws + take((size_t)NE * sizeof(float)));

    const int B = 256;
    auto nb = [&](long long n) { return (unsigned)((n + B - 1) / B); };

    // ---- weight conversion to bf16 ----
    cvt_f32_bf16<<<nb(F_IN * H1D), B, 0, stream>>>(W1, W1b, F_IN * H1D);
    cvt_f32_bf16<<<nb(H1D * H2D),  B, 0, stream>>>(W2, W2b, H1D * H2D);
    cvt_f32_bf16<<<nb(H1D * H2D),  B, 0, stream>>>(W3, W3b, H1D * H2D);

    // ================= Layer 1: h1 = GAT(x, W1, a1), ReLU ==================
    {
        dim3 g((NN + 127) / 128, H1D / 16);
        wmma_gemm_bf16<false, H1D><<<g, B, (size_t)F_IN * 32, stream>>>(
            x, W1b, Wh1, NN, F_IN);
    }
    node_st<H1D><<<nb((long long)NN * 32), B, 0, stream>>>(Wh1, a1, sBuf, tBuf, NN);
    fill_u32<<<nb(NN), B, 0, stream>>>(menc, NEG_INF_ENC, NN);
    fill_f32<<<nb(NN), B, 0, stream>>>(denom, 0.f, NN);
    fill_f32<<<nb((long long)NN * H1D), B, 0, stream>>>(hp1, 0.f, NN * H1D);
    edge_logits_max<<<nb(NE), B, 0, stream>>>(src, dst, sBuf, tBuf, ework, menc, NE);
    node_max_fix<<<nb(NN), B, 0, stream>>>(menc, mBuf, NN);
    edge_exp_sum<<<nb(NE), B, 0, stream>>>(src, mBuf, ework, denom, NE);
    edge_scatter<H1D><<<nb((long long)NE * 32), B, 0, stream>>>(
        src, dst, ework, denom, Wh1, hp1, NE);

    // ============ Layers 2 & 3: mu / logvar = GAT(relu(h1), ...) ===========
    {
        dim3 g((NN + 127) / 128, H2D / 16);
        wmma_gemm_bf16<true, H2D><<<g, B, (size_t)H1D * 32, stream>>>(
            hp1, W2b, Wh2, NN, H1D);
        wmma_gemm_bf16<true, H2D><<<g, B, (size_t)H1D * 32, stream>>>(
            hp1, W3b, Wh3, NN, H1D);
    }

    float* zOut  = out;
    float* muOut = out + (size_t)NN * H2D;
    float* lvOut = out + (size_t)2 * NN * H2D;

    // --- layer 2 (mu) ---
    node_st<H2D><<<nb((long long)NN * 32), B, 0, stream>>>(Wh2, a2, sBuf, tBuf, NN);
    fill_u32<<<nb(NN), B, 0, stream>>>(menc, NEG_INF_ENC, NN);
    fill_f32<<<nb(NN), B, 0, stream>>>(denom, 0.f, NN);
    fill_f32<<<nb((long long)NN * H2D), B, 0, stream>>>(muOut, 0.f, NN * H2D);
    edge_logits_max<<<nb(NE), B, 0, stream>>>(src, dst, sBuf, tBuf, ework, menc, NE);
    node_max_fix<<<nb(NN), B, 0, stream>>>(menc, mBuf, NN);
    edge_exp_sum<<<nb(NE), B, 0, stream>>>(src, mBuf, ework, denom, NE);
    edge_scatter<H2D><<<nb((long long)NE * 32), B, 0, stream>>>(
        src, dst, ework, denom, Wh2, muOut, NE);

    // --- layer 3 (logvar) ---
    node_st<H2D><<<nb((long long)NN * 32), B, 0, stream>>>(Wh3, a3, sBuf, tBuf, NN);
    fill_u32<<<nb(NN), B, 0, stream>>>(menc, NEG_INF_ENC, NN);
    fill_f32<<<nb(NN), B, 0, stream>>>(denom, 0.f, NN);
    fill_f32<<<nb((long long)NN * H2D), B, 0, stream>>>(lvOut, 0.f, NN * H2D);
    edge_logits_max<<<nb(NE), B, 0, stream>>>(src, dst, sBuf, tBuf, ework, menc, NE);
    node_max_fix<<<nb(NN), B, 0, stream>>>(menc, mBuf, NN);
    edge_exp_sum<<<nb(NE), B, 0, stream>>>(src, mBuf, ework, denom, NE);
    edge_scatter<H2D><<<nb((long long)NE * 32), B, 0, stream>>>(
        src, dst, ework, denom, Wh3, lvOut, NE);

    // z = mu (eval mode reparameterize)
    copy_f32<<<nb((long long)NN * H2D), B, 0, stream>>>(muOut, zOut, NN * H2D);
}
